// NT_Xent_13950053778105
// MI455X (gfx1250) — matrix operand order repaired and checked by reading, not compiled, not run on hardware
//
#include <hip/hip_runtime.h>
#include <stdint.h>

#define TEMP_INV 2.0f                     // 1 / TEMPERATURE
#define LOG2E    1.4426950408889634f
#define LN2      0.6931471805599453f
#define B_ROWS   4096
#define NTOT     8192
#define D        512
#define NCHUNK   8
#define CHUNK_COLS (NTOT / NCHUNK)        // 1024
#define NTILES   (CHUNK_COLS / 16)        // 64
#define MBLOCK   256                      // rows per workgroup (8 waves x 32)
#define LDS_ROW_H 520                     // halfs per LDS tile row (512 + 8 pad)

typedef __attribute__((ext_vector_type(16))) __bf16 v16bf;
typedef __attribute__((ext_vector_type(8)))  __bf16 v8bf;
typedef __attribute__((ext_vector_type(8)))  float  v8f;

__device__ __forceinline__ unsigned int pack_bf16(float a, float b) {
  union { float f; unsigned int u; } ua, ub;
  ua.f = a; ub.f = b;
  unsigned int x = (ua.u + 0x7FFFu + ((ua.u >> 16) & 1u)) >> 16;   // RNE
  unsigned int y = (ub.u + 0x7FFFu + ((ub.u >> 16) & 1u)) >> 16;
  return x | (y << 16);
}

__device__ __forceinline__ v16bf cat8(v8bf lo, v8bf hi) {
  return __builtin_shufflevector(lo, hi, 0,1,2,3,4,5,6,7,8,9,10,11,12,13,14,15);
}

// ---------------- kernel 1: L2-normalize rows of [z_i; z_j] -> bf16 P[8192][512]
__global__ __launch_bounds__(256) void nk_normalize(
    const float* __restrict__ zi, const float* __restrict__ zj,
    unsigned short* __restrict__ pbf)
{
  int wave = threadIdx.x >> 5;
  int lane = threadIdx.x & 31;
  int row  = blockIdx.x * 8 + wave;                 // one wave per row
  const float* src = (row < B_ROWS) ? (zi + (size_t)row * D)
                                    : (zj + (size_t)(row - B_ROWS) * D);
  const float4* s4 = (const float4*)(src + lane * 16);
  float4 f[4];
  float ss = 0.0f;
  #pragma unroll
  for (int i = 0; i < 4; ++i) {
    f[i] = s4[i];
    ss += f[i].x*f[i].x + f[i].y*f[i].y + f[i].z*f[i].z + f[i].w*f[i].w;
  }
  #pragma unroll
  for (int off = 1; off < 32; off <<= 1) ss += __shfl_xor(ss, off, 32);
  float rn = __builtin_amdgcn_rsqf(ss);

  unsigned int w[8];
  #pragma unroll
  for (int i = 0; i < 4; ++i) {
    w[2*i]   = pack_bf16(f[i].x * rn, f[i].y * rn);
    w[2*i+1] = pack_bf16(f[i].z * rn, f[i].w * rn);
  }
  uint4* dst = (uint4*)(pbf + (size_t)row * D + lane * 16);
  dst[0] = make_uint4(w[0], w[1], w[2], w[3]);
  dst[1] = make_uint4(w[4], w[5], w[6], w[7]);
}

// Stage one 16-row x 512-half B tile into LDS via async DMA (4 x b128 per thread).
__device__ __forceinline__ void stage_tile(const unsigned short* __restrict__ pbf,
                                           int cbase, unsigned int ldsBase, int tid) {
  #pragma unroll
  for (int i = 0; i < 4; ++i) {
    int idx = tid + i * 256;                        // 1024 x 16B chunks
    int r = idx >> 6;
    int c = idx & 63;
    unsigned long long gaddr = (unsigned long long)(uintptr_t)
        (pbf + (size_t)(cbase + r) * D + c * 8);
    unsigned int laddr = ldsBase + r * (LDS_ROW_H * 2) + c * 16;
    asm volatile("global_load_async_to_lds_b128 %0, %1, off"
                 :: "v"(laddr), "v"(gaddr) : "memory");
  }
}

// ---------------- kernel 2: fused P.P^T/T + online masked logsumexp + pos capture
// Each wave owns 32 rows (2 x 16-row tiles) so every B fragment feeds 2 WMMAs.
__global__ __launch_bounds__(256) void nk_simloss(
    const unsigned short* __restrict__ pbf,
    float* __restrict__ outM, float* __restrict__ outS, float* __restrict__ outP)
{
  __shared__ unsigned short tileB[2][16 * LDS_ROW_H];   // double buffer, 33 KB
  const int tid  = threadIdx.x;
  const int wave = tid >> 5;
  const int lane = tid & 31;
  const int hv   = lane >> 4;                        // lane half (0/1)
  const int l16  = lane & 15;
  const int chunk   = blockIdx.x & (NCHUNK - 1);
  const int mblk    = blockIdx.x >> 3;
  const int rowBase = mblk * MBLOCK + wave * 32;
  const int colStart = chunk * CHUNK_COLS;

  // Preload A fragments for both 16-row tiles (2 x 16 x v16bf = 256 VGPRs).
  v16bf afrag[2][16];
  #pragma unroll
  for (int s = 0; s < 2; ++s) {
    const unsigned short* arow = pbf + (size_t)(rowBase + s * 16 + l16) * D;
    #pragma unroll
    for (int kt = 0; kt < 16; ++kt) {
      const unsigned short* ap = arow + kt * 32 + hv * 16;
      afrag[s][kt] = cat8(*(const v8bf*)ap, *(const v8bf*)(ap + 8));
    }
  }

  float mx[2][8], sm[2][8], ps[2][8];
  #pragma unroll
  for (int s = 0; s < 2; ++s)
    #pragma unroll
    for (int e = 0; e < 8; ++e) { mx[s][e] = -64.0f; sm[s][e] = 0.0f; ps[s][e] = 0.0f; }

  const unsigned int ldsB0 = (unsigned int)(uintptr_t)(&tileB[0][0]);
  const unsigned int ldsB1 = (unsigned int)(uintptr_t)(&tileB[1][0]);

  stage_tile(pbf, colStart, ldsB0, tid);             // prologue: tile 0 -> buf 0

  for (int nt = 0; nt < NTILES; ++nt) {
    const int sel = nt & 1;
    if (nt + 1 < NTILES) {                           // overlap DMA with compute
      stage_tile(pbf, colStart + (nt + 1) * 16, sel ? ldsB0 : ldsB1, tid);
      asm volatile("s_wait_asynccnt 0x4" ::: "memory");  // in-order: tile nt landed
    } else {
      asm volatile("s_wait_asynccnt 0x0" ::: "memory");
    }
    __syncthreads();

    const int cbase = colStart + nt * 16;
    const unsigned short* brow = &tileB[sel][l16 * LDS_ROW_H];

    v8f accA = {0.f,0.f,0.f,0.f,0.f,0.f,0.f,0.f};
    v8f accB = {0.f,0.f,0.f,0.f,0.f,0.f,0.f,0.f};
    // Software-pipelined K loop: prefetch B fragment kt+1 while 2 WMMAs run on kt.
    const unsigned short* bp0 = brow;
    v16bf bc = cat8(*(const v8bf*)(bp0 + hv * 16), *(const v8bf*)(bp0 + hv * 16 + 8));
    #pragma unroll
    for (int kt = 0; kt < 16; ++kt) {
      v16bf bn;
      if (kt < 15) {
        const unsigned short* bp = brow + (kt + 1) * 32 + hv * 16;
        bn = cat8(*(const v8bf*)bp, *(const v8bf*)(bp + 8));
      }
      accA = __builtin_amdgcn_wmma_f32_16x16x32_bf16(
          false, afrag[0][kt], false, bc, (short)0, accA, false, false);
      accB = __builtin_amdgcn_wmma_f32_16x16x32_bf16(
          false, afrag[1][kt], false, bc, (short)0, accB, false, false);
      bc = bn;
    }

    // Per-slot online softmax (log2 domain) + positive/diagonal handling.
    const int n = cbase + l16;
    #pragma unroll
    for (int s = 0; s < 2; ++s) {
      const v8f& acc = s ? accB : accA;
      #pragma unroll
      for (int e = 0; e < 8; ++e) {
        int m = rowBase + s * 16 + e + hv * 8;
        float logit2 = acc[e] * (TEMP_INV * LOG2E);
        ps[s][e] += (n == ((m + B_ROWS) & (NTOT - 1))) ? acc[e] * TEMP_INV : 0.0f;
        float v = (m == n) ? -1.0e30f : logit2;      // mask self-similarity
        float nm = fmaxf(mx[s][e], v);
        sm[s][e] = sm[s][e] * __builtin_amdgcn_exp2f(mx[s][e] - nm)
                 + __builtin_amdgcn_exp2f(v - nm);
        mx[s][e] = nm;
      }
    }
    __syncthreads();
  }

  // Merge the 16 column-lanes of each row.
  #pragma unroll
  for (int s = 0; s < 2; ++s) {
    #pragma unroll
    for (int e = 0; e < 8; ++e) {
      #pragma unroll
      for (int off = 1; off < 16; off <<= 1) {
        float om = __shfl_xor(mx[s][e], off, 32);
        float os = __shfl_xor(sm[s][e], off, 32);
        float op = __shfl_xor(ps[s][e], off, 32);
        float nm = fmaxf(mx[s][e], om);
        sm[s][e] = sm[s][e] * __builtin_amdgcn_exp2f(mx[s][e] - nm)
                 + os       * __builtin_amdgcn_exp2f(om - nm);
        mx[s][e] = nm;
        ps[s][e] += op;
      }
    }
  }
  if (l16 == 0) {
    #pragma unroll
    for (int s = 0; s < 2; ++s) {
      #pragma unroll
      for (int e = 0; e < 8; ++e) {
        int m = rowBase + s * 16 + e + hv * 8;
        size_t o = (size_t)chunk * NTOT + m;
        outM[o] = mx[s][e]; outS[o] = sm[s][e]; outP[o] = ps[s][e];
      }
    }
  }
}

// ---------------- kernel 3: deterministic merge of N-chunks + final mean
__global__ __launch_bounds__(1024) void nk_finalize(
    const float* __restrict__ outM, const float* __restrict__ outS,
    const float* __restrict__ outP, float* __restrict__ out)
{
  __shared__ float red[1024];
  const int tid = threadIdx.x;
  float local = 0.0f;
  #pragma unroll
  for (int t = 0; t < NTOT / 1024; ++t) {
    int m = tid + t * 1024;
    float M = outM[m], S = outS[m], P = outP[m];
    #pragma unroll
    for (int c = 1; c < NCHUNK; ++c) {
      float Mc = outM[(size_t)c * NTOT + m];
      float Sc = outS[(size_t)c * NTOT + m];
      float nm = fmaxf(M, Mc);
      S = S * __builtin_amdgcn_exp2f(M - nm) + Sc * __builtin_amdgcn_exp2f(Mc - nm);
      M = nm;
      P += outP[(size_t)c * NTOT + m];
    }
    float lse = (M + __builtin_amdgcn_logf(S)) * LN2;  // back to natural log
    local += lse - P;
  }
  red[tid] = local;
  __syncthreads();
  for (int s = 512; s > 0; s >>= 1) {
    if (tid < s) red[tid] += red[tid + s];
    __syncthreads();
  }
  if (tid == 0) out[0] = red[0] * (1.0f / (float)NTOT);
}

extern "C" void kernel_launch(void* const* d_in, const int* in_sizes, int n_in,
                              void* d_out, int out_size, void* d_ws, size_t ws_size,
                              hipStream_t stream) {
  (void)in_sizes; (void)n_in; (void)out_size; (void)ws_size;
  const float* zi = (const float*)d_in[0];
  const float* zj = (const float*)d_in[1];
  float* out = (float*)d_out;

  unsigned short* pbf = (unsigned short*)d_ws;                 // 8 MB bf16 P
  size_t pbytes = (size_t)NTOT * D * sizeof(unsigned short);
  float* outM = (float*)((char*)d_ws + pbytes);                // [NCHUNK][NTOT]
  float* outS = outM + (size_t)NCHUNK * NTOT;
  float* outP = outS + (size_t)NCHUNK * NTOT;

  nk_normalize<<<NTOT / 8, 256, 0, stream>>>(zi, zj, pbf);
  nk_simloss<<<(NTOT / MBLOCK) * NCHUNK, 256, 0, stream>>>(pbf, outM, outS, outP);
  nk_finalize<<<1, 1024, 0, stream>>>(outM, outS, outP, out);
}
